// AuxFormer_38173669327302
// MI455X (gfx1250) — matrix operand order repaired and checked by reading, not compiled, not run on hardware
//
#include <hip/hip_runtime.h>

#define BATCH 4
#define NPTS  8192
#define NBRS  17          // K=16 neighbors + self
#define EPSV  1e-5f

typedef __attribute__((ext_vector_type(16))) _Float16     v16h;
typedef __attribute__((ext_vector_type(8)))  float        v8f;
typedef __attribute__((ext_vector_type(8)))  unsigned int v8u;

__device__ __forceinline__ unsigned int pack2h(float a, float b) {
  union { _Float16 h[2]; unsigned int u; } p;
  p.h[0] = (_Float16)a; p.h[1] = (_Float16)b;
  return p.u;
}

// =====================================================================
// Generic WMMA GEMM:  C[M,N] = act( A[M,K] @ W[K,N] + bias )
// f32 -> f16 staged through LDS, f32 accumulation via v_wmma_f32_16x16x32_f16
// Block tile 64x32, 8 waves (wave32), each wave owns one 16x16 C tile.
// A staged row-major (K contiguous), B staged column-major (K contiguous)
// so both fragments fill with 8x ds_load_b32; stride 34 halves = 17 banks
// (odd) -> conflict-free, and keeps every paired load 4B-aligned.
// =====================================================================
#define AS_STRIDE 34
#define BS_STRIDE 34

__global__ __launch_bounds__(256) void k_gemm(const float* __restrict__ A,
                                              const float* __restrict__ W,
                                              const float* __restrict__ bias,
                                              float* __restrict__ C,
                                              int M, int Kd, int Nd, int relu) {
  __shared__ _Float16 As[64 * AS_STRIDE];   // [row][k]
  __shared__ _Float16 Bs[32 * BS_STRIDE];   // [col][k]
  const int tid    = threadIdx.x;
  const int lane   = tid & 31;
  const int wid    = tid >> 5;
  const int warpMO = (wid & 3) << 4;   // 0..48
  const int warpNO = (wid >> 2) << 4;  // 0,16
  const int blockM = blockIdx.y << 6;
  const int blockN = blockIdx.x << 5;
  const int laneHi = lane >> 4;        // 0 or 1
  const int lmod   = lane & 15;

  const bool aFull = (blockM + 64 <= M);
  const bool bFull = (blockN + 32 <= Nd);

  v8f acc = {};
  for (int k0 = 0; k0 < Kd; k0 += 32) {
    const bool kFull = (k0 + 32 <= Kd);

    // ---- stage A tile (64 rows x 32 k) ----
    if (aFull && kFull) {
      // vector fast path: 512 float4 loads, packed cvt, 32-bit LDS stores
#pragma unroll
      for (int it = 0; it < 2; ++it) {
        int q  = tid + it * 256;           // 0..511
        int r  = q >> 3;                   // row 0..63
        int c4 = q & 7;                    // float4 index in row
        const float4 v = *(const float4*)(A + (size_t)(blockM + r) * Kd + k0 + c4 * 4);
        *(unsigned int*)&As[r * AS_STRIDE + c4 * 4]     = pack2h(v.x, v.y);
        *(unsigned int*)&As[r * AS_STRIDE + c4 * 4 + 2] = pack2h(v.z, v.w);
      }
    } else {
      for (int t = tid; t < 64 * 32; t += 256) {
        int r = t >> 5, c = t & 31;
        int gr = blockM + r, gc = k0 + c;
        float v = (gr < M && gc < Kd) ? A[(size_t)gr * Kd + gc] : 0.0f;
        As[r * AS_STRIDE + c] = (_Float16)v;
      }
    }
    // ---- stage B tile (32 k x 32 cols), column-major pairs ----
    if (bFull && kFull) {
#pragma unroll
      for (int it = 0; it < 2; ++it) {
        int q  = tid + it * 256;           // 0..511 -> (k-pair, col)
        int r2 = q >> 5;                   // k pair 0..15
        int c  = q & 31;                   // col
        float v0 = W[(size_t)(k0 + 2 * r2) * Nd + blockN + c];
        float v1 = W[(size_t)(k0 + 2 * r2 + 1) * Nd + blockN + c];
        *(unsigned int*)&Bs[c * BS_STRIDE + 2 * r2] = pack2h(v0, v1);
      }
    } else {
      for (int t = tid; t < 32 * 32; t += 256) {
        int r = t >> 5, c = t & 31;
        int gr = k0 + r, gc = blockN + c;
        float v = (gr < Kd && gc < Nd) ? W[(size_t)gr * Nd + gc] : 0.0f;
        Bs[c * BS_STRIDE + r] = (_Float16)v;
      }
    }
    // speculative prefetch of next K tile (global_prefetch_b8)
    if (k0 + 32 < Kd && tid < 64) {
      int gr = blockM + tid;
      if (gr < M) __builtin_prefetch(&A[(size_t)gr * Kd + k0 + 32], 0, 1);
    }
    __syncthreads();

    // A fragment (ISA 16-bit 16x32 layout): lanes 0-15 K-base 0, 16-31 K-base +8;
    // VGPRs 0-3 cover K 0..7(+base), VGPRs 4-7 cover K 16..23(+base).
    const int mrow = warpMO + lmod;
    const int kbA  = laneHi * 8;
    v8u au;
#pragma unroll
    for (int i = 0; i < 8; ++i) {
      int kk = ((i < 4) ? (2 * i) : (16 + 2 * (i - 4))) + kbA;
      au[i] = *(const unsigned int*)&As[mrow * AS_STRIDE + kk];
    }
    // B fragment: lane holds column N=lane%16; same K layout as A (lanes 16-31 K-base +8)
    const int ncol = warpNO + lmod;
    const int kbB  = laneHi * 8;
    v8u bu;
#pragma unroll
    for (int i = 0; i < 8; ++i)
      bu[i] = *(const unsigned int*)&Bs[ncol * BS_STRIDE + kbB + ((i < 4) ? (2 * i) : (16 + 2 * (i - 4)))];

    v16h afr = __builtin_bit_cast(v16h, au);
    v16h bfr = __builtin_bit_cast(v16h, bu);
    acc = __builtin_amdgcn_wmma_f32_16x16x32_f16(false, afr, false, bfr,
                                                 (short)0, acc, false, false);
    __syncthreads();
  }
  // C/D layout: VGPR r -> M = r + (laneHi? 8 : 0), N = lane%16
#pragma unroll
  for (int r = 0; r < 8; ++r) {
    int mrow = blockM + warpMO + r + laneHi * 8;
    int ncol = blockN + warpNO + lmod;
    if (mrow < M && ncol < Nd) {
      float v = acc[r] + (bias ? bias[ncol] : 0.0f);
      if (relu) v = fmaxf(v, 0.0f);
      C[(size_t)mrow * Nd + ncol] = v;
    }
  }
}

// =====================================================================
// Elementwise / reduction helper kernels
// =====================================================================
__global__ void k_zero(float* p, long n) {
  long t = (long)blockIdx.x * blockDim.x + threadIdx.x;
  if (t < n) p[t] = 0.0f;
}

__global__ void k_slice(const float* __restrict__ data, float* __restrict__ pos,
                        float* __restrict__ aux, int total) {
  int t = blockIdx.x * blockDim.x + threadIdx.x;
  if (t >= total) return;
  const float* d = data + (size_t)t * 7;
  pos[t * 3 + 0] = d[0];  pos[t * 3 + 1] = d[1];  pos[t * 3 + 2] = d[2];
  aux[t * 4 + 0] = d[3];  aux[t * 4 + 1] = d[4];
  aux[t * 4 + 2] = d[5];  aux[t * 4 + 3] = d[6];
}

__global__ void k_iota(int* cum, int total, int n) {
  int t = blockIdx.x * blockDim.x + threadIdx.x;
  if (t < total) cum[t] = t % n;
}

// BatchNorm stats: per-channel sum / sumsq via LDS partials + global atomics
__global__ void k_bn_stats(const float* __restrict__ X, int M, int C,
                           float* __restrict__ stats) {
  __shared__ float ssum[512];
  __shared__ float ssq[512];
  for (int c = threadIdx.x; c < C; c += blockDim.x) { ssum[c] = 0.f; ssq[c] = 0.f; }
  __syncthreads();
  int r0 = blockIdx.x * 256;
  int total = 256 * C;
  for (int t = threadIdx.x; t < total; t += blockDim.x) {
    int r = r0 + t / C, c = t % C;
    if (r < M) {
      float v = X[(size_t)r * C + c];
      atomicAdd(&ssum[c], v);
      atomicAdd(&ssq[c], v * v);
    }
  }
  __syncthreads();
  for (int c = threadIdx.x; c < C; c += blockDim.x) {
    atomicAdd(&stats[c], ssum[c]);
    atomicAdd(&stats[C + c], ssq[c]);
  }
}

__global__ void k_bn_apply(const float* __restrict__ X, const float* __restrict__ stats,
                           const float* __restrict__ g, const float* __restrict__ be,
                           float* __restrict__ Y, long M, int C, int relu) {
  long t = (long)blockIdx.x * blockDim.x + threadIdx.x;
  if (t >= M * C) return;
  int c = (int)(t % C);
  float mean = stats[c] / (float)M;
  float var  = stats[C + c] / (float)M - mean * mean;
  float v = (X[t] - mean) * rsqrtf(var + EPSV) * g[c] + be[c];
  if (relu) v = fmaxf(v, 0.0f);
  Y[t] = v;
}

// LayerNorm(residual): y = LN(x + r) * g + be   (one thread per row)
__global__ void k_ln_res(const float* __restrict__ x, const float* __restrict__ r,
                         const float* __restrict__ g, const float* __restrict__ be,
                         float* __restrict__ y, int P, int d) {
  int row = blockIdx.x * blockDim.x + threadIdx.x;
  if (row >= P) return;
  size_t base = (size_t)row * d;
  float s = 0.f;
  for (int c = 0; c < d; ++c) s += x[base + c] + r[base + c];
  float mean = s / d;
  float v = 0.f;
  for (int c = 0; c < d; ++c) {
    float t = x[base + c] + r[base + c] - mean;
    v += t * t;
  }
  float rs = rsqrtf(v / d + EPSV);
  for (int c = 0; c < d; ++c)
    y[base + c] = (x[base + c] + r[base + c] - mean) * rs * g[c] + be[c];
}

// =====================================================================
// KNN: per-thread query, top-16 insertion list, ref points staged in LDS
// grid = ((mq+127)/128, B)
// =====================================================================
__global__ __launch_bounds__(128) void k_knn(const float* __restrict__ qpos,
                                             const float* __restrict__ rpos,
                                             int* __restrict__ out,
                                             int mq, int nr, int kout,
                                             int excl, int appendSelf) {
  __shared__ float tile[128 * 3];
  const int b  = blockIdx.y;
  const int qi = blockIdx.x * 128 + threadIdx.x;
  const bool valid = (qi < mq);
  float qx = 0, qy = 0, qz = 0;
  if (valid) {
    const float* q = qpos + ((size_t)b * mq + qi) * 3;
    qx = q[0]; qy = q[1]; qz = q[2];
  }
  float bd[16];
  int   bi[16];
#pragma unroll
  for (int t = 0; t < 16; ++t) { bd[t] = 3.0e38f; bi[t] = 0; }

  for (int j0 = 0; j0 < nr; j0 += 128) {
    int cnt = min(128, nr - j0);
    for (int t = threadIdx.x; t < cnt * 3; t += 128)
      tile[t] = rpos[((size_t)b * nr + j0) * 3 + t];
    __syncthreads();
    if (valid) {
      for (int jl = 0; jl < cnt; ++jl) {
        int j = j0 + jl;
        if (excl && j == qi) continue;
        float dx = qx - tile[jl * 3 + 0];
        float dy = qy - tile[jl * 3 + 1];
        float dz = qz - tile[jl * 3 + 2];
        float d = dx * dx + dy * dy + dz * dz;
        if (d < bd[15]) {
          bd[15] = d; bi[15] = j;
#pragma unroll
          for (int t = 15; t > 0; --t) {
            if (bd[t] < bd[t - 1]) {
              float tf = bd[t]; bd[t] = bd[t - 1]; bd[t - 1] = tf;
              int   ti = bi[t]; bi[t] = bi[t - 1]; bi[t - 1] = ti;
            }
          }
        }
      }
    }
    __syncthreads();
  }
  if (valid) {
    int* o = out + ((size_t)b * mq + qi) * kout;
#pragma unroll
    for (int t = 0; t < 16; ++t) o[t] = bi[t];
    if (appendSelf) o[16] = qi;
  }
}

// =====================================================================
// Farthest-point sampling: one block per batch (matches scan semantics:
// idx[t] = last BEFORE update, last_{t+1} = argmax of updated dmin)
// =====================================================================
__global__ __launch_bounds__(256) void k_fps(const float* __restrict__ pos,
                                             int n, int m, int* __restrict__ out) {
  __shared__ float smax[256];
  __shared__ int   simax[256];
  __shared__ float spx, spy, spz;
  const int b = blockIdx.x;
  const float* P = pos + (size_t)b * n * 3;
  const int tid = threadIdx.x;
  float dm[32];
#pragma unroll
  for (int c = 0; c < 32; ++c) dm[c] = 3.0e38f;
  int last = 0;
  for (int t = 0; t < m; ++t) {
    if (tid == 0) {
      out[(size_t)b * m + t] = last;
      spx = P[last * 3 + 0]; spy = P[last * 3 + 1]; spz = P[last * 3 + 2];
    }
    __syncthreads();
    float bmax = -1.0f; int bidx = 0x7fffffff;
#pragma unroll
    for (int c = 0; c < 32; ++c) {
      int i = c * 256 + tid;
      if (i < n) {
        float dx = P[i * 3 + 0] - spx;
        float dy = P[i * 3 + 1] - spy;
        float dz = P[i * 3 + 2] - spz;
        float d = dx * dx + dy * dy + dz * dz;
        dm[c] = fminf(dm[c], d);
        if (dm[c] > bmax || (dm[c] == bmax && i < bidx)) { bmax = dm[c]; bidx = i; }
      }
    }
    smax[tid] = bmax; simax[tid] = bidx;
    __syncthreads();
    for (int s2 = 128; s2 > 0; s2 >>= 1) {
      if (tid < s2) {
        if (smax[tid + s2] > smax[tid] ||
            (smax[tid + s2] == smax[tid] && simax[tid + s2] < simax[tid])) {
          smax[tid] = smax[tid + s2]; simax[tid] = simax[tid + s2];
        }
      }
      __syncthreads();
    }
    last = simax[0];
    __syncthreads();
  }
}

// =====================================================================
// Gather / scatter helpers
// =====================================================================
__global__ void k_gather_rows(const float* __restrict__ in, const int* __restrict__ idx,
                              float* __restrict__ out, int m, int nin, int C) {
  long t = (long)blockIdx.x * blockDim.x + threadIdx.x;
  long total = (long)BATCH * m * C;
  if (t >= total) return;
  int b = (int)(t / ((long)m * C));
  long rem = t % ((long)m * C);
  int i = (int)(rem / C), c = (int)(rem % C);
  int src = idx[(size_t)b * m + i];
  out[((size_t)b * m + i) * C + c] = in[((size_t)b * nin + src) * C + c];
}

__global__ void k_gather_ints(const int* __restrict__ cold, const int* __restrict__ idx,
                              int* __restrict__ cnew, int m, int nin) {
  int t = blockIdx.x * blockDim.x + threadIdx.x;
  if (t >= BATCH * m) return;
  int b = t / m;
  cnew[t] = cold[(size_t)b * nin + idx[t]];
}

__global__ void k_gather_max(const float* __restrict__ xt, const int* __restrict__ nbr,
                             float* __restrict__ out, int m, int nin, int C) {
  long t = (long)blockIdx.x * blockDim.x + threadIdx.x;
  long total = (long)BATCH * m * C;
  if (t >= total) return;
  int b = (int)(t / ((long)m * C));
  long rem = t % ((long)m * C);
  int i = (int)(rem / C), c = (int)(rem % C);
  const int* nb = nbr + ((size_t)b * m + i) * 16;
  float best = -3.0e38f;
#pragma unroll
  for (int j = 0; j < 16; ++j) {
    float v = xt[((size_t)b * nin + nb[j]) * C + c];
    best = fmaxf(best, v);
  }
  out[((size_t)b * m + i) * C + c] = best;
}

// =====================================================================
// Point-transformer conv helpers
// =====================================================================
__global__ void k_pairdiff(const float* __restrict__ pos, const int* __restrict__ nbr,
                           float* __restrict__ diff, int Pn, int nloc) {
  long t = (long)blockIdx.x * blockDim.x + threadIdx.x;
  long total = (long)Pn * NBRS;
  if (t >= total) return;
  int p = (int)(t / NBRS);
  int b = p / nloc;
  int nb = nbr[t];
  const float* pi = pos + (size_t)p * 3;
  const float* pj = pos + ((size_t)b * nloc + nb) * 3;
  diff[t * 3 + 0] = pi[0] - pj[0];
  diff[t * 3 + 1] = pi[1] - pj[1];
  diff[t * 3 + 2] = pi[2] - pj[2];
}

__global__ void k_pair_attn(const float* __restrict__ adst, const float* __restrict__ asrc,
                            const float* __restrict__ delta, const int* __restrict__ nbr,
                            float* __restrict__ attn, int Pn, int nloc, int d) {
  long t = (long)blockIdx.x * blockDim.x + threadIdx.x;
  long total = (long)Pn * NBRS * d;
  if (t >= total) return;
  long pj = t / d;
  int c = (int)(t % d);
  int p = (int)(pj / NBRS);
  int b = p / nloc;
  int nb = nbr[pj];
  attn[t] = adst[(size_t)p * d + c] - asrc[((size_t)b * nloc + nb) * d + c] + delta[t];
}

// softmax over 17 neighbors (per channel) + weighted sum of (val_j + delta_j)
__global__ void k_ptc_reduce(const float* __restrict__ alpha, const float* __restrict__ val,
                             const float* __restrict__ delta, const int* __restrict__ nbr,
                             float* __restrict__ out, int Pn, int nloc, int d) {
  long t = (long)blockIdx.x * blockDim.x + threadIdx.x;
  long total = (long)Pn * d;
  if (t >= total) return;
  int p = (int)(t / d);
  int c = (int)(t % d);
  int b = p / nloc;
  size_t base = (size_t)p * NBRS * d + c;
  float mx = -3.0e38f;
#pragma unroll
  for (int j = 0; j < NBRS; ++j) mx = fmaxf(mx, alpha[base + (size_t)j * d]);
  float se = 0.f, acc = 0.f;
#pragma unroll
  for (int j = 0; j < NBRS; ++j) {
    float e = expf(alpha[base + (size_t)j * d] - mx);
    se += e;
    int nb = nbr[(size_t)p * NBRS + j];
    acc += e * (val[((size_t)b * nloc + nb) * d + c] + delta[base + (size_t)j * d]);
  }
  out[t] = acc / se;
}

// cross-attention gate: vg = v * sigmoid(dot(q,k)/sqrt(d)) (one thread per row)
__global__ void k_gate(const float* __restrict__ q, const float* __restrict__ k,
                       const float* __restrict__ v, float* __restrict__ vg,
                       int P, int d) {
  int row = blockIdx.x * blockDim.x + threadIdx.x;
  if (row >= P) return;
  size_t base = (size_t)row * d;
  float dot = 0.f;
  for (int c = 0; c < d; ++c) dot += q[base + c] * k[base + c];
  float g = 1.f / (1.f + expf(-dot * rsqrtf((float)d)));
  for (int c = 0; c < d; ++c) vg[base + c] = v[base + c] * g;
}

// pooling helpers
__global__ void k_snr_w(const float* __restrict__ aux, const int* __restrict__ cum,
                        float* __restrict__ w, int nfin) {
  int t = blockIdx.x * blockDim.x + threadIdx.x;
  if (t >= BATCH * nfin) return;
  int b = t / nfin;
  float s = aux[((size_t)b * NPTS + cum[t]) * 4 + 2];
  w[t] = fmaxf(s, 0.0f);
}

__global__ void k_pool(const float* __restrict__ x, const float* __restrict__ w,
                       float* __restrict__ out, int nfin, int C) {
  int t = blockIdx.x * blockDim.x + threadIdx.x;
  if (t >= BATCH * C) return;
  int b = t / C, c = t % C;
  float sw = 0.f, acc = 0.f;
  for (int i = 0; i < nfin; ++i) {
    float wv = w[b * nfin + i];
    sw  += wv;
    acc += x[((size_t)b * nfin + i) * C + c] * wv;
  }
  out[t] = acc / fmaxf(sw, 1e-6f);
}

// =====================================================================
// Host orchestration
// =====================================================================
extern "C" void kernel_launch(void* const* d_in, const int* in_sizes, int n_in,
                              void* d_out, int out_size, void* d_ws, size_t ws_size,
                              hipStream_t stream) {
  (void)in_sizes; (void)n_in; (void)out_size; (void)ws_size;
  const int DIMS[5] = {32, 64, 128, 256, 512};
  const int BN_ = BATCH * NPTS;

  // --- param indices: jax tree_flatten (sorted dict keys) of setup_inputs() ---
  // 0: data
  // aux_heads[s]: W=1+2s, b=2+2s  (s=0..4)                     -> 1..10
  // aux_trunk: W1=11 W2=12 b1=13 b2=14 be1=15 be2=16 g1=17 g2=18
  // head: W1=19 W2=20 b1=21 b2=22
  // mlp_input: W=23 b=24 be=25 g=26
  // td[i]: base=27+4i: W,b,be,g                                 -> 27..42
  // tr[i] (tb): base=43+15i: Wdst,Wlin,Wsrc, aW1,aW2,ab1,ab2,
  //             liW,lib, loW,lob, pW1,pW2,pb1,pb2               -> 43..102
  // tr0 (tb): base=103                                          -> 103..117
  // xattn[i]: base=118+10i: Wk,Wo,Wq,Wv, be,bk,bo,bq,bv,g       -> 118..167
  const float* data = (const float*)d_in[0];
  auto Pm = [&](int i) { return (const float*)d_in[i]; };

  // --- workspace arena ---
  char* wsb = (char*)d_ws;
  size_t off = 0;
  auto allocF = [&](size_t n) { float* p = (float*)(wsb + off); off += ((n * 4 + 255) / 256) * 256; return p; };
  auto allocI = [&](size_t n) { int*   p = (int*)(wsb + off);   off += ((n * 4 + 255) / 256) * 256; return p; };

  float* POS0  = allocF((size_t)BN_ * 3);
  float* AUX0  = allocF((size_t)BN_ * 4);
  float* T1    = allocF((size_t)BN_ * 64);
  float* T2    = allocF((size_t)BN_ * 64);
  float* AUXTOK = allocF((size_t)BN_ * (32 + 64 + 128 + 256 + 512));
  size_t tokoff[5]; { size_t o = 0; for (int s = 0; s < 5; ++s) { tokoff[s] = o; o += (size_t)BN_ * DIMS[s]; } }
  float* XA    = allocF((size_t)BN_ * 32);
  float* XB    = allocF((size_t)BN_ * 32);
  float* X1    = allocF((size_t)BN_ * 32);
  float* VAL   = allocF((size_t)BN_ * 32);
  float* ASRC  = allocF((size_t)BN_ * 32);
  float* ADST  = allocF((size_t)BN_ * 32);
  float* DIFF  = allocF((size_t)BN_ * NBRS * 3);
  float* Hbuf  = allocF((size_t)BN_ * NBRS * 64);   // biggest scratch (pos/attn hidden, bn tmp, xattn out)
  float* DELTA = allocF((size_t)BN_ * NBRS * 32);
  float* ATTN  = allocF((size_t)BN_ * NBRS * 32);
  float* XT    = allocF((size_t)BN_ * 64);
  float* AUXF  = allocF((size_t)1 << 20);
  float* STATS = allocF(1024);
  float* WPOOL = allocF(BATCH * 32);
  float* POOLED = allocF(BATCH * 512);
  float* H2    = allocF(BATCH * 64);
  float* POSA  = allocF((size_t)BATCH * 2048 * 3);
  float* POSB  = allocF((size_t)BATCH * 2048 * 3);
  int* NBR    = allocI((size_t)BN_ * NBRS);
  int* NBRTD  = allocI((size_t)BATCH * 2048 * 16);
  int* FPSIDX = allocI((size_t)BATCH * 2048);
  int* CUMA   = allocI(BN_);
  int* CUMB   = allocI(BN_);

  auto gemm = [&](const float* A, const float* W, const float* bias, float* C,
                  int M, int K, int N, int relu) {
    dim3 g((N + 31) / 32, (M + 63) / 64);
    k_gemm<<<g, 256, 0, stream>>>(A, W, bias, C, M, K, N, relu);
  };
  auto bnApply = [&](const float* Xin, const float* g, const float* be,
                     float* Y, int M, int C, int relu) {
    k_zero<<<(2 * C + 255) / 256, 256, 0, stream>>>(STATS, 2 * C);
    k_bn_stats<<<(M + 255) / 256, 256, 0, stream>>>(Xin, M, C, STATS);
    long tot = (long)M * C;
    k_bn_apply<<<(unsigned)((tot + 255) / 256), 256, 0, stream>>>(Xin, STATS, g, be, Y, M, C, relu);
  };
  auto tblock = [&](const float* xin, const float* pos, const int* nbr,
                    int Pn, int nloc, int d, int bi, float* xout) {
    long P17 = (long)Pn * NBRS;
    gemm(xin, Pm(bi + 7), Pm(bi + 8), X1, Pn, d, d, 1);          // lin_in + relu
    gemm(X1, Pm(bi + 1), nullptr, VAL,  Pn, d, d, 0);            // Wlin
    gemm(X1, Pm(bi + 2), nullptr, ASRC, Pn, d, d, 0);            // Wsrc
    gemm(X1, Pm(bi + 0), nullptr, ADST, Pn, d, d, 0);            // Wdst
    k_pairdiff<<<(unsigned)((P17 + 255) / 256), 256, 0, stream>>>(pos, nbr, DIFF, Pn, nloc);
    gemm(DIFF, Pm(bi + 11), Pm(bi + 13), Hbuf,  (int)P17, 3,  64, 1);  // pos_nn L1
    gemm(Hbuf, Pm(bi + 12), Pm(bi + 14), DELTA, (int)P17, 64, d,  1);  // pos_nn L2
    long tpd = P17 * d;
    k_pair_attn<<<(unsigned)((tpd + 255) / 256), 256, 0, stream>>>(ADST, ASRC, DELTA, nbr, ATTN, Pn, nloc, d);
    gemm(ATTN, Pm(bi + 3), Pm(bi + 5), Hbuf, (int)P17, d,  64, 1);     // attn_nn L1
    gemm(Hbuf, Pm(bi + 4), Pm(bi + 6), ATTN, (int)P17, 64, d,  1);     // attn_nn L2 -> alpha
    long tpc = (long)Pn * d;
    k_ptc_reduce<<<(unsigned)((tpc + 255) / 256), 256, 0, stream>>>(ATTN, VAL, DELTA, nbr, ADST, Pn, nloc, d);
    gemm(ADST, Pm(bi + 9), Pm(bi + 10), xout, Pn, d, d, 1);      // lin_out + relu
  };
  auto xattn = [&](const float* xin, const float* auxf, int Pn, int d, int bi, float* xout) {
    gemm(xin,  Pm(bi + 2), Pm(bi + 7), VAL,  Pn, d, d, 0);       // q
    gemm(auxf, Pm(bi + 0), Pm(bi + 5), ASRC, Pn, d, d, 0);       // k
    gemm(auxf, Pm(bi + 3), Pm(bi + 8), ADST, Pn, d, d, 0);       // v
    k_gate<<<(Pn + 255) / 256, 256, 0, stream>>>(VAL, ASRC, ADST, DELTA, Pn, d);
    gemm(DELTA, Pm(bi + 1), Pm(bi + 6), Hbuf, Pn, d, d, 0);      // Wo
    k_ln_res<<<(Pn + 255) / 256, 256, 0, stream>>>(xin, Hbuf, Pm(bi + 9), Pm(bi + 4), xout, Pn, d);
  };

  // ---------------- forward pass ----------------
  k_slice<<<(BN_ + 255) / 256, 256, 0, stream>>>(data, POS0, AUX0, BN_);

  float* x = XA; float* xalt = XB;
  // mlp_input: relu(bn(pos @ W + b))
  gemm(POS0, Pm(23), Pm(24), Hbuf, BN_, 3, 32, 0);
  bnApply(Hbuf, Pm(26), Pm(25), x, BN_, 32, 1);

  // tr0 on full resolution
  k_knn<<<dim3((NPTS + 127) / 128, BATCH), 128, 0, stream>>>(POS0, POS0, NBR, NPTS, NPTS, NBRS, 1, 1);
  tblock(x, POS0, NBR, BN_, NPTS, 32, 103, xalt);
  { float* t = x; x = xalt; xalt = t; }

  // aux trunk + heads
  gemm(AUX0, Pm(11), Pm(13), Hbuf, BN_, 4, 64, 0);
  bnApply(Hbuf, Pm(17), Pm(15), T1, BN_, 64, 1);
  gemm(T1, Pm(12), Pm(14), Hbuf, BN_, 64, 64, 0);
  bnApply(Hbuf, Pm(18), Pm(16), T2, BN_, 64, 1);
  for (int s = 0; s < 5; ++s)
    gemm(T2, Pm(1 + 2 * s), Pm(2 + 2 * s), AUXTOK + tokoff[s], BN_, 64, DIMS[s], 0);

  // xattn level 0
  xattn(x, AUXTOK + tokoff[0], BN_, 32, 118, xalt);
  { float* t = x; x = xalt; xalt = t; }

  k_iota<<<(BN_ + 255) / 256, 256, 0, stream>>>(CUMA, BN_, NPTS);
  int* cum = CUMA; int* cumAlt = CUMB;
  const float* curpos = POS0;
  float* posbuf[2] = {POSA, POSB}; int pb = 0;
  int n = NPTS;

  for (int s = 0; s < 4; ++s) {
    int m = n / 4;
    int din = DIMS[s], dout = DIMS[s + 1];
    // FPS + subsample positions
    k_fps<<<BATCH, 256, 0, stream>>>(curpos, n, m, FPSIDX);
    float* spos = posbuf[pb]; pb ^= 1;
    { long tot = (long)BATCH * m * 3;
      k_gather_rows<<<(unsigned)((tot + 255) / 256), 256, 0, stream>>>(curpos, FPSIDX, spos, m, n, 3); }
    // transition-down KNN (sub -> full), then bn-linear + max-pool gather
    k_knn<<<dim3((m + 127) / 128, BATCH), 128, 0, stream>>>(spos, curpos, NBRTD, m, n, 16, 0, 0);
    gemm(x, Pm(27 + 4 * s), Pm(27 + 4 * s + 1), Hbuf, BATCH * n, din, dout, 0);
    bnApply(Hbuf, Pm(27 + 4 * s + 3), Pm(27 + 4 * s + 2), XT, BATCH * n, dout, 1);
    { long tot = (long)BATCH * m * dout;
      k_gather_max<<<(unsigned)((tot + 255) / 256), 256, 0, stream>>>(XT, NBRTD, xalt, m, n, dout); }
    { float* t = x; x = xalt; xalt = t; }
    curpos = spos;
    k_gather_ints<<<(BATCH * m + 255) / 256, 256, 0, stream>>>(cum, FPSIDX, cumAlt, m, n);
    { int* t = cum; cum = cumAlt; cumAlt = t; }
    // self neighbors + transformer block at new level
    k_knn<<<dim3((m + 127) / 128, BATCH), 128, 0, stream>>>(spos, spos, NBR, m, m, NBRS, 1, 1);
    tblock(x, spos, NBR, BATCH * m, m, dout, 43 + 15 * s, xalt);
    { float* t = x; x = xalt; xalt = t; }
    // cross-attention with gathered aux tokens
    { long tot = (long)BATCH * m * dout;
      k_gather_rows<<<(unsigned)((tot + 255) / 256), 256, 0, stream>>>(AUXTOK + tokoff[s + 1], cum, AUXF, m, NPTS, dout); }
    xattn(x, AUXF, BATCH * m, dout, 118 + 10 * (s + 1), xalt);
    { float* t = x; x = xalt; xalt = t; }
    n = m;
  }

  // weighted pooling + head -> d_out (B x 49)
  k_snr_w<<<(BATCH * n + 255) / 256, 256, 0, stream>>>(AUX0, cum, WPOOL, n);
  k_pool<<<(BATCH * 512 + 255) / 256, 256, 0, stream>>>(x, WPOOL, POOLED, n, 512);
  gemm(POOLED, Pm(19), Pm(21), H2, BATCH, 512, 64, 1);
  gemm(H2, Pm(20), Pm(22), (float*)d_out, BATCH, 64, 49, 0);
}
